// RegionProposalNetwork_23854248362771
// MI455X (gfx1250) — matrix operand order repaired and checked by reading, not compiled, not run on hardware
//
#include <hip/hip_runtime.h>
#include <hip/hip_bf16.h>
#include <math.h>

typedef __attribute__((ext_vector_type(16))) _Float16 v16h;
typedef __attribute__((ext_vector_type(8)))  _Float16 v8h;
typedef __attribute__((ext_vector_type(8)))  float    v8f;
typedef unsigned long long u64;
typedef unsigned int       u32;

#define GH 160
#define GW 160
#define NPOS (GH*GW)        // 25600 spatial positions
#define CIN 256
#define NA 9
#define NTOT (NPOS*NA)      // 230400 anchors
#define KTOT (9*CIN)        // 2304 GEMM K
#define IMG 2560.0f
#define ANCHOR_STRIDE 16.0f
#define PRENMS 6000
#define NPAD1 262144        // 2^18 >= 230400
#define NPAD2 8192          // 2^13 >= 6000
#define POSTK 300
#define NMS_THR 0.7f
#define DWH_CLAMP 4.135166556742356f  // log(1000/16)

#define WAVES_PER_WG 8
#define BROWS 256           // B rows staged per k-step (= all out channels)
#define BSTRIDE 40          // halves per LDS row: 32 data + 8 pad (80 B, bank-friendly)
#define LO_OFF (BROWS*BSTRIDE)

// ---------------------------------------------------------------------------
// Prep: feat NCHW f32 -> NHWC f16 hi/lo split;  conv_w OIHW f32 -> B[N][K] hi/lo
// ---------------------------------------------------------------------------
__global__ void prep_feat(const float* __restrict__ feat,
                          _Float16* __restrict__ fHi, _Float16* __restrict__ fLo) {
  int i = blockIdx.x * blockDim.x + threadIdx.x;
  if (i >= CIN * NPOS) return;
  int c = i / NPOS, pos = i % NPOS;
  float v = feat[i];
  _Float16 h = (_Float16)v;
  float r = v - (float)h;
  fHi[(size_t)pos * CIN + c] = h;
  fLo[(size_t)pos * CIN + c] = (_Float16)r;
}

__global__ void prep_w(const float* __restrict__ wsrc,
                       _Float16* __restrict__ bHi, _Float16* __restrict__ bLo) {
  int i = blockIdx.x * blockDim.x + threadIdx.x;
  if (i >= 256 * KTOT) return;
  int n = i / KTOT, k = i % KTOT;
  int tap = k / CIN, c = k % CIN;           // K layout: k = tap*256 + c_in
  float v = wsrc[((size_t)n * CIN + c) * 9 + tap];
  _Float16 h = (_Float16)v;
  bHi[i] = h;
  bLo[i] = (_Float16)(v - (float)h);
}

// ---------------------------------------------------------------------------
// 3x3 conv + bias + ReLU as implicit GEMM on v_wmma_f32_16x16x32_f16.
// Workgroup = 8 waves x 16-position M tiles; B (hi+lo) for the current
// 32-wide K slab is staged once per WG into padded LDS with
// global_load_async_to_lds_b128 (ASYNCcnt), then each wave reads its
// fragments from LDS. Split precision: acc += aH*bH + aH*bL + aL*bH.
// ---------------------------------------------------------------------------
__global__ void __launch_bounds__(256) conv3_wmma(
    const _Float16* __restrict__ fHi, const _Float16* __restrict__ fLo,
    const _Float16* __restrict__ bHi, const _Float16* __restrict__ bLo,
    const float* __restrict__ bias, float* __restrict__ rpn) {
  __shared__ _Float16 sB[2 * BROWS * BSTRIDE];   // 40 KB: [hi | lo]

  const int tid  = threadIdx.x;
  const int lane = tid & 31;
  const int wv   = tid >> 5;
  const int m0   = (blockIdx.x * WAVES_PER_WG + wv) << 4;
  const int row  = lane & 15;          // A row / B column within tile
  const int grp  = lane >> 4;          // lane group (K-phase)
  const int p    = m0 + row;
  const int py   = p / GW, px = p % GW;

  // async-copy ownership: thread t stages B row n=t (hi and lo), 4x16B each
  const u64 gHiRow = (u64)(uintptr_t)(bHi + (size_t)tid * KTOT);
  const u64 gLoRow = (u64)(uintptr_t)(bLo + (size_t)tid * KTOT);
  const u32 ldsHi  = (u32)(uintptr_t)(&sB[tid * BSTRIDE]);
  const u32 ldsLo  = (u32)(uintptr_t)(&sB[LO_OFF + tid * BSTRIDE]);

  v8f acc[16] = {};

  for (int tap = 0; tap < 9; ++tap) {
    const int dy = tap / 3 - 1, dx = tap % 3 - 1;
    const int sy = py + dy, sx = px + dx;
    const bool valid = (sy >= 0) && (sy < GH) && (sx >= 0) && (sx < GW);
    const _Float16* aHiRow = fHi + (size_t)(sy * GW + sx) * CIN;
    const _Float16* aLoRow = fLo + (size_t)(sy * GW + sx) * CIN;

    for (int c0 = 0; c0 < CIN; c0 += 32) {
      const int koff = tap * CIN + c0;               // k offset of this slab

      // ---- stage B slab (32 K x 256 N, hi+lo) into LDS asynchronously ----
      __syncthreads();                                // previous slab fully read
      {
        const u64 ga = gHiRow + (u64)koff * 2;
        const u64 gb = gLoRow + (u64)koff * 2;
#pragma unroll
        for (int part = 0; part < 4; ++part) {
          asm volatile("global_load_async_to_lds_b128 %0, %1, off"
                       :: "v"(ldsHi + part * 16), "v"(ga + (u64)(part * 16))
                       : "memory");
          asm volatile("global_load_async_to_lds_b128 %0, %1, off"
                       :: "v"(ldsLo + part * 16), "v"(gb + (u64)(part * 16))
                       : "memory");
        }
        asm volatile("s_wait_asynccnt 0x0" ::: "memory");
      }
      __syncthreads();                                // slab visible to all waves

      // ---- A fragment (16x32 f16) straight from global ----
      v16h aH = {}, aL = {};
      if (valid) {
        const int o0 = c0 + grp * 16;
        const int o1 = c0 + grp * 16 + 8;
        v8h h0 = *(const v8h*)(aHiRow + o0);
        v8h h1 = *(const v8h*)(aHiRow + o1);
        v8h l0 = *(const v8h*)(aLoRow + o0);
        v8h l1 = *(const v8h*)(aLoRow + o1);
#pragma unroll
        for (int e = 0; e < 8; ++e) {
          aH[e] = h0[e]; aH[8 + e] = h1[e];
          aL[e] = l0[e]; aL[8 + e] = l1[e];
        }
      }

      // ---- 16 N tiles from LDS ----
#pragma unroll
      for (int nt = 0; nt < 16; ++nt) {
        const int n = (nt << 4) + row;                // B column = out channel
        const v16h bH = *(const v16h*)(sB + (size_t)n * BSTRIDE + grp * 16);
        const v16h bL = *(const v16h*)(sB + LO_OFF + (size_t)n * BSTRIDE + grp * 16);
        acc[nt] = __builtin_amdgcn_wmma_f32_16x16x32_f16(false, aL, false, bH,
                                                         (short)0, acc[nt], false, false);
        acc[nt] = __builtin_amdgcn_wmma_f32_16x16x32_f16(false, aH, false, bL,
                                                         (short)0, acc[nt], false, false);
        acc[nt] = __builtin_amdgcn_wmma_f32_16x16x32_f16(false, aH, false, bH,
                                                         (short)0, acc[nt], false, false);
      }
    }
  }

  // C/D layout: lane N = lane&15; VGPR r -> M = r + 8*(lane>>4)
#pragma unroll
  for (int nt = 0; nt < 16; ++nt) {
    const int n = (nt << 4) + row;
    const float bv = bias[n];
#pragma unroll
    for (int r = 0; r < 8; ++r) {
      const int M = r + (grp << 3);
      float v = acc[nt][r] + bv;
      rpn[(size_t)(m0 + M) * CIN + n] = v > 0.0f ? v : 0.0f;
    }
  }
}

// ---------------------------------------------------------------------------
// 1x1 cls/bbox heads + anchor generation + box decode + sigmoid, one block/pos
// ---------------------------------------------------------------------------
__global__ void __launch_bounds__(64) head_decode(
    const float* __restrict__ rpn,
    const float* __restrict__ cw, const float* __restrict__ cb,
    const float* __restrict__ bw, const float* __restrict__ bb,
    float* __restrict__ scores, float* __restrict__ props) {
  __shared__ float sf[CIN];
  __shared__ float sdelt[36];
  __shared__ float scls[NA];
  const int pos = blockIdx.x;
  const int tid = threadIdx.x;
  for (int i = tid; i < CIN; i += 64) sf[i] = rpn[(size_t)pos * CIN + i];
  __syncthreads();
  if (tid < 45) {
    const float* wrow;
    float bv;
    if (tid < NA) { wrow = cw + tid * CIN; bv = cb[tid]; }
    else          { wrow = bw + (tid - NA) * CIN; bv = bb[tid - NA]; }
    float s = 0.0f;
    for (int i = 0; i < CIN; ++i) s = fmaf(sf[i], wrow[i], s);
    s += bv;
    if (tid < NA) scls[tid] = s; else sdelt[tid - NA] = s;
  }
  __syncthreads();
  if (tid < NA) {
    const int a = tid;
    const int ar_i = a / 3, s_i = a % 3;               // a = ar_idx*3 + scale_idx
    const float scl = (s_i == 0) ? 128.0f : ((s_i == 1) ? 256.0f : 512.0f);
    const float ar  = (ar_i == 0) ? 0.5f : ((ar_i == 1) ? 1.0f : 2.0f);
    const float hr = sqrtf(ar), wr = 1.0f / hr;
    const float wsz = wr * scl, hsz = hr * scl;
    const float x1b = rintf(-wsz * 0.5f), y1b = rintf(-hsz * 0.5f);
    const float x2b = rintf( wsz * 0.5f), y2b = rintf( hsz * 0.5f);
    const float shx = (float)(pos % GW) * ANCHOR_STRIDE;
    const float shy = (float)(pos / GW) * ANCHOR_STRIDE;
    const float ax1 = shx + x1b, ay1 = shy + y1b, ax2 = shx + x2b, ay2 = shy + y2b;
    const float w = ax2 - ax1, h = ay2 - ay1;
    const float cx = ax1 + 0.5f * w, cy = ay1 + 0.5f * h;
    const float ddx = sdelt[a * 4 + 0], ddy = sdelt[a * 4 + 1];
    const float ddw = fminf(sdelt[a * 4 + 2], DWH_CLAMP);
    const float ddh = fminf(sdelt[a * 4 + 3], DWH_CLAMP);
    const float pcx = ddx * w + cx, pcy = ddy * h + cy;
    const float pw = expf(ddw) * w, ph = expf(ddh) * h;
    const size_t o = ((size_t)pos * NA + a) * 4;
    props[o + 0] = pcx - 0.5f * pw;
    props[o + 1] = pcy - 0.5f * ph;
    props[o + 2] = pcx + 0.5f * pw;
    props[o + 3] = pcy + 0.5f * ph;
    scores[(size_t)pos * NA + a] = 1.0f / (1.0f + expf(-scls[a]));
  }
}

// ---------------------------------------------------------------------------
// Exact top-k via full bitonic sort on (monotone score bits, ~index) keys
// ---------------------------------------------------------------------------
__device__ __forceinline__ u32 fmono(float s) {
  u32 b = __float_as_uint(s);
  return b ^ ((b & 0x80000000u) ? 0xFFFFFFFFu : 0x80000000u);
}

__global__ void build_keys1(const float* __restrict__ scores, u64* __restrict__ keys) {
  int i = blockIdx.x * blockDim.x + threadIdx.x;
  if (i >= NPAD1) return;
  keys[i] = (i < NTOT) ? (((u64)fmono(scores[i]) << 32) | (u32)(~(u32)i)) : 0ull;
}

__global__ void bitonic_pass(u64* __restrict__ keys, u32 j, u32 k) {
  u32 i = blockIdx.x * blockDim.x + threadIdx.x;
  u32 ixj = i ^ j;
  if (ixj > i) {
    u64 a = keys[i], b = keys[ixj];
    bool up = ((i & k) == 0);                 // descending order overall
    if (up ? (a < b) : (a > b)) { keys[i] = b; keys[ixj] = a; }
  }
}

__global__ void gather_topk(const u64* __restrict__ keys, const float* __restrict__ scores,
                            const float* __restrict__ props,
                            float* __restrict__ boxes, float* __restrict__ tsc,
                            int* __restrict__ keep) {
  int t = blockIdx.x * blockDim.x + threadIdx.x;
  if (t >= PRENMS) return;
  u32 idx = ~(u32)keys[t];
  float x1 = fminf(fmaxf(props[(size_t)idx * 4 + 0], 0.0f), IMG);
  float y1 = fminf(fmaxf(props[(size_t)idx * 4 + 1], 0.0f), IMG);
  float x2 = fminf(fmaxf(props[(size_t)idx * 4 + 2], 0.0f), IMG);
  float y2 = fminf(fmaxf(props[(size_t)idx * 4 + 3], 0.0f), IMG);
  boxes[t * 4 + 0] = x1; boxes[t * 4 + 1] = y1;
  boxes[t * 4 + 2] = x2; boxes[t * 4 + 3] = y2;
  tsc[t] = scores[idx];
  keep[t] = ((x2 - x1) >= 16.0f) && ((y2 - y1) >= 16.0f);
}

// ---------------------------------------------------------------------------
// Greedy NMS: one workgroup, keep flags in LDS, boxes stream from L2
// ---------------------------------------------------------------------------
__global__ void __launch_bounds__(1024) nms_kernel(const float* __restrict__ boxes,
                                                   int* __restrict__ keep) {
  __shared__ unsigned char kp[PRENMS];
  const int tid = threadIdx.x;
  for (int t = tid; t < PRENMS; t += 1024) kp[t] = (unsigned char)keep[t];
  __syncthreads();
  for (int i = 0; i < PRENMS - 1; ++i) {
    __syncthreads();
    if (!kp[i]) continue;                       // uniform (LDS value after barrier)
    const float x1 = boxes[i * 4 + 0], y1 = boxes[i * 4 + 1];
    const float x2 = boxes[i * 4 + 2], y2 = boxes[i * 4 + 3];
    const float ai = (x2 - x1) * (y2 - y1);
    for (int j = i + 1 + tid; j < PRENMS; j += 1024) {
      if (!kp[j]) continue;
      const float bx1 = boxes[j * 4 + 0], by1 = boxes[j * 4 + 1];
      const float bx2 = boxes[j * 4 + 2], by2 = boxes[j * 4 + 3];
      const float xl = fmaxf(x1, bx1), yt = fmaxf(y1, by1);
      const float xr = fminf(x2, bx2), yb = fminf(y2, by2);
      const float inter = fmaxf(xr - xl, 0.0f) * fmaxf(yb - yt, 0.0f);
      const float aj = (bx2 - bx1) * (by2 - by1);
      const float iou = inter / (ai + aj - inter);
      if (iou > NMS_THR) kp[j] = 0;
    }
  }
  __syncthreads();
  for (int t = tid; t < PRENMS; t += 1024) keep[t] = kp[t];
}

__global__ void build_keys2(const float* __restrict__ tsc, const int* __restrict__ keep,
                            float* __restrict__ fs, u64* __restrict__ keys2) {
  int t = blockIdx.x * blockDim.x + threadIdx.x;
  if (t >= NPAD2) return;
  if (t < PRENMS) {
    float s = keep[t] ? tsc[t] : -1.0f;
    fs[t] = s;
    keys2[t] = ((u64)fmono(s) << 32) | (u32)(~(u32)t);
  } else {
    keys2[t] = 0ull;
  }
}

__global__ void write_out(const u64* __restrict__ keys2, const float* __restrict__ boxes,
                          const float* __restrict__ fs, float* __restrict__ out) {
  int t = blockIdx.x * blockDim.x + threadIdx.x;
  if (t >= POSTK) return;
  u32 t6 = ~(u32)keys2[t];
  out[t * 5 + 0] = boxes[t6 * 4 + 0];
  out[t * 5 + 1] = boxes[t6 * 4 + 1];
  out[t * 5 + 2] = boxes[t6 * 4 + 2];
  out[t * 5 + 3] = boxes[t6 * 4 + 3];
  out[t * 5 + 4] = fs[t6];
}

// ---------------------------------------------------------------------------
extern "C" void kernel_launch(void* const* d_in, const int* in_sizes, int n_in,
                              void* d_out, int out_size, void* d_ws, size_t ws_size,
                              hipStream_t stream) {
  (void)in_sizes; (void)n_in; (void)out_size; (void)ws_size;
  const float* feat  = (const float*)d_in[0];
  // d_in[1] = image (zeros, only shape matters -> constants above)
  const float* convw = (const float*)d_in[2];
  const float* convb = (const float*)d_in[3];
  const float* clsw  = (const float*)d_in[4];
  const float* clsb  = (const float*)d_in[5];
  const float* bbw   = (const float*)d_in[6];
  const float* bbb   = (const float*)d_in[7];
  float* out = (float*)d_out;

  char* base = (char*)d_ws;
  size_t off = 0;
  auto alloc = [&](size_t bytes) -> void* {
    off = (off + 255) & ~(size_t)255;
    void* p = base + off;
    off += bytes;
    return p;
  };
  _Float16* fHi   = (_Float16*)alloc((size_t)NPOS * CIN * 2);   // 13.1 MB
  _Float16* fLo   = (_Float16*)alloc((size_t)NPOS * CIN * 2);   // 13.1 MB
  _Float16* bHi   = (_Float16*)alloc((size_t)256 * KTOT * 2);   // 1.18 MB
  _Float16* bLo   = (_Float16*)alloc((size_t)256 * KTOT * 2);   // 1.18 MB
  float*    rpn   = (float*)alloc((size_t)NPOS * CIN * 4);      // 26.2 MB
  float*    sco   = (float*)alloc((size_t)NTOT * 4);            // 0.92 MB
  float*    props = (float*)alloc((size_t)NTOT * 4 * 4);        // 3.69 MB
  u64*      keys  = (u64*)alloc((size_t)NPAD1 * 8);             // 2.10 MB
  float*    box6  = (float*)alloc((size_t)PRENMS * 4 * 4);
  float*    tsc   = (float*)alloc((size_t)PRENMS * 4);
  int*      keep  = (int*)alloc((size_t)PRENMS * 4);
  float*    fs    = (float*)alloc((size_t)PRENMS * 4);
  u64*      keys2 = (u64*)alloc((size_t)NPAD2 * 8);

  // 1) precision-split prep
  prep_feat<<<(CIN * NPOS + 255) / 256, 256, 0, stream>>>(feat, fHi, fLo);
  prep_w<<<(256 * KTOT + 255) / 256, 256, 0, stream>>>(convw, bHi, bLo);

  // 2) 3x3 conv via WMMA implicit GEMM: 200 WGs x 8 waves, B staged in LDS
  conv3_wmma<<<NPOS / (16 * WAVES_PER_WG), 32 * WAVES_PER_WG, 0, stream>>>(
      fHi, fLo, bHi, bLo, convb, rpn);

  // 3) heads + anchors + decode + sigmoid
  head_decode<<<NPOS, 64, 0, stream>>>(rpn, clsw, clsb, bbw, bbb, sco, props);

  // 4) top-6000 via bitonic sort of 2^18 keys
  build_keys1<<<NPAD1 / 256, 256, 0, stream>>>(sco, keys);
  for (u32 k = 2; k <= NPAD1; k <<= 1)
    for (u32 j = k >> 1; j > 0; j >>= 1)
      bitonic_pass<<<NPAD1 / 256, 256, 0, stream>>>(keys, j, k);

  // 5) gather + clip + min-size
  gather_topk<<<(PRENMS + 255) / 256, 256, 0, stream>>>(keys, sco, props, box6, tsc, keep);

  // 6) greedy NMS (single workgroup)
  nms_kernel<<<1, 1024, 0, stream>>>(box6, keep);

  // 7) top-300 of final scores via bitonic sort of 2^13 keys
  build_keys2<<<NPAD2 / 256, 256, 0, stream>>>(tsc, keep, fs, keys2);
  for (u32 k = 2; k <= NPAD2; k <<= 1)
    for (u32 j = k >> 1; j > 0; j >>= 1)
      bitonic_pass<<<NPAD2 / 256, 256, 0, stream>>>(keys2, j, k);

  // 8) emit 300 x [x1,y1,x2,y2,score]
  write_out<<<(POSTK + 255) / 256, 256, 0, stream>>>(keys2, box6, fs, out);
}
